// VAE_42872363548834
// MI455X (gfx1250) — matrix operand order repaired and checked by reading, not compiled, not run on hardware
//
#include <hip/hip_runtime.h>
#include <math.h>

// ---------------- sizes ----------------
#define S_   256
#define B_   64
#define I_   64
#define H_   256
#define D2_  512
#define G4H  1024
#define LAT_ 64
#define ATT_ 64

typedef __attribute__((ext_vector_type(16))) _Float16 v16h;
typedef __attribute__((ext_vector_type(8)))  float    v8f;

// ---------------- WMMA fragment helpers (CDNA5 wave32 layouts) ----------------
// A (16x32 f16, row-major source, leading dim lda):
//   lanes 0-15: M=lane,  K = [0..7]  in v0..v3, [16..23] in v4..v7
//   lanes16-31: M=lane-16, K = [8..15], [24..31]
__device__ __forceinline__ v16h load_a_frag(const _Float16* A, int lda, int m0, int k0) {
  const int lane = threadIdx.x & 31;
  const _Float16* p = A + (size_t)(m0 + (lane & 15)) * lda + k0 + ((lane >> 4) << 3);
  v16h r;
  reinterpret_cast<float4*>(&r)[0] = *reinterpret_cast<const float4*>(p);
  reinterpret_cast<float4*>(&r)[1] = *reinterpret_cast<const float4*>(p + 16);
  return r;
}
// B (32x16 f16). Weights stored [N, K] (PyTorch layout = B^T), so lane n reads
// K values of output-feature row n0+(lane&15), same K order as A:
//   lanes 0-15 hold K=[0..7],[16..23]; lanes 16-31 hold K=[8..15],[24..31].
__device__ __forceinline__ v16h load_b_frag(const _Float16* W, int ldw, int n0, int k0) {
  const int lane = threadIdx.x & 31;
  const _Float16* p = W + (size_t)(n0 + (lane & 15)) * ldw + k0 + ((lane >> 4) << 3);
  v16h r;
  reinterpret_cast<float4*>(&r)[0] = *reinterpret_cast<const float4*>(p);
  reinterpret_cast<float4*>(&r)[1] = *reinterpret_cast<const float4*>(p + 16);
  return r;
}

__device__ __forceinline__ v8f wmma32(v16h a, v16h b, v8f c) {
  return __builtin_amdgcn_wmma_f32_16x16x32_f16(false, a, false, b, (short)0, c,
                                                false, false);
}

__device__ __forceinline__ float sigmoidf_(float x) { return 1.f / (1.f + __expf(-x)); }
__device__ __forceinline__ float softplusf_(float x) {
  return (x > 20.f) ? x : log1pf(__expf(x));
}

// ---------------- generic WMMA GEMM: C[M,N] = A[M,K] * W[N,K]^T + bias ----------------
// out_mode: 0 = f32 linear [M,N], 1 = f16 linear [M,N],
//           2 = f16 fragment-swizzled [tile][lane][8] (16B/lane, coalesced)
__global__ __launch_bounds__(256) void wmma_gemm_kernel(
    const _Float16* __restrict__ A, const _Float16* __restrict__ W,
    const float* __restrict__ bias, void* __restrict__ C,
    int M, int N, int K, int out_mode)
{
  const int lane = threadIdx.x & 31;
  const int wave = threadIdx.x >> 5;
  const int ntn  = N >> 4;
  const int tile = blockIdx.x * (blockDim.x >> 5) + wave;
  if (tile >= (M >> 4) * ntn) return;
  const int mt = tile / ntn, nt = tile % ntn;

  v8f acc = {0.f,0.f,0.f,0.f,0.f,0.f,0.f,0.f};
  for (int k0 = 0; k0 < K; k0 += 32) {
    v16h a = load_a_frag(A, K, mt * 16, k0);
    v16h b = load_b_frag(W, K, nt * 16, k0);
    acc = wmma32(a, b, acc);
  }
  const int ncol  = nt * 16 + (lane & 15);
  const int mbase = mt * 16 + ((lane >> 4) << 3);
  const float bv  = bias ? bias[ncol] : 0.f;
  if (out_mode == 2) {
    // fragment-native store: one 16B write per lane
    _Float16 tmp[8];
#pragma unroll
    for (int r = 0; r < 8; ++r) tmp[r] = (_Float16)(acc[r] + bv);
    _Float16* Ch = (_Float16*)C + ((size_t)tile * 32 + lane) * 8;
    *reinterpret_cast<float4*>(Ch) = *reinterpret_cast<const float4*>(tmp);
  } else if (out_mode == 1) {
    _Float16* Ch = (_Float16*)C;
#pragma unroll
    for (int r = 0; r < 8; ++r)
      Ch[(size_t)(mbase + r) * N + ncol] = (_Float16)(acc[r] + bv);
  } else {
    float* Cf = (float*)C;
#pragma unroll
    for (int r = 0; r < 8; ++r)
      Cf[(size_t)(mbase + r) * N + ncol] = acc[r] + bv;
  }
}

// ---------------- persistent bidirectional LSTM recurrence ----------------
// One block per direction; 32 waves. h kept in LDS (f16), c kept in registers
// (each wave exclusively owns two 16x16 output tiles across all 4 gates).
// xW = precomputed x@Wih^T + b in fragment-swizzled f16 layout (out_mode 2):
// tiles are (s*4 + mt) * 64 + (g*16 + nt), 32 lanes * 8 f16 each.
// Gate order i,f,g,o.
__global__ __launch_bounds__(1024) void lstm_bidir_kernel(
    const _Float16* __restrict__ xW_f, const _Float16* __restrict__ xW_b,
    const _Float16* __restrict__ Whh_f, const _Float16* __restrict__ Whh_b,
    _Float16* __restrict__ hs_f, _Float16* __restrict__ hs_b,
    float* __restrict__ hT_f, float* __restrict__ hT_b)
{
  const int dir = blockIdx.x;
  const _Float16* xW  = dir ? xW_b  : xW_f;
  const _Float16* Whh = dir ? Whh_b : Whh_f;
  _Float16* hs = dir ? hs_b : hs_f;
  float*    hT = dir ? hT_b : hT_f;

  __shared__ __align__(16) _Float16 h_lds[B_ * H_];   // 32 KB

  for (int i = threadIdx.x; i < B_ * H_; i += 1024) h_lds[i] = (_Float16)0.f;
  __syncthreads();

  const int lane = threadIdx.x & 31;
  const int wave = threadIdx.x >> 5;
  const int p0   = wave * 2;            // two adjacent (mt,nt) positions, same mt
  const int mt   = p0 >> 4;
  const int nt0  = p0 & 15;
  const int hi8  = (lane >> 4) << 3;
  const int lc   = lane & 15;

  float creg[2][8];
#pragma unroll
  for (int p = 0; p < 2; ++p)
#pragma unroll
    for (int r = 0; r < 8; ++r) creg[p][r] = 0.f;

  for (int t = 0; t < S_; ++t) {
    const int s = dir ? (S_ - 1 - t) : t;

    // ---- accumulator init from swizzled xW: 8 x b128 loads per wave ----
    const size_t tbase = (size_t)(s * 4 + mt) * 64;   // 64 N-tiles per M-tile row
    v8f acc[2][4];
#pragma unroll
    for (int p = 0; p < 2; ++p) {
#pragma unroll
      for (int g = 0; g < 4; ++g) {
        const size_t tl = tbase + (size_t)(g * 16 + nt0 + p);
        const _Float16* src = xW + (tl * 32 + lane) * 8;
        _Float16 tmp[8];
        *reinterpret_cast<float4*>(tmp) = *reinterpret_cast<const float4*>(src);
        v8f a;
#pragma unroll
        for (int r = 0; r < 8; ++r) a[r] = (float)tmp[r];
        acc[p][g] = a;
      }
    }
    // ---- gates += h @ Whh^T : 8 WMMAs per K-slice, A-frag shared ----
#pragma unroll 1
    for (int kk = 0; kk < H_; kk += 32) {
      v16h afrag = load_a_frag(h_lds, H_, mt * 16, kk);
#pragma unroll
      for (int p = 0; p < 2; ++p) {
        const int nt = nt0 + p;
#pragma unroll
        for (int g = 0; g < 4; ++g) {
          v16h bfrag = load_b_frag(Whh, H_, g * H_ + nt * 16, kk);
          acc[p][g] = wmma32(afrag, bfrag, acc[p][g]);
        }
      }
    }
    // ---- LSTM cell (c in registers, exclusively owned) ----
    float hnew[2][8];
#pragma unroll
    for (int p = 0; p < 2; ++p) {
#pragma unroll
      for (int r = 0; r < 8; ++r) {
        float ig = sigmoidf_(acc[p][0][r]);
        float fg = sigmoidf_(acc[p][1][r]);
        float gg = tanhf(acc[p][2][r]);
        float og = sigmoidf_(acc[p][3][r]);
        float c  = fg * creg[p][r] + ig * gg;
        creg[p][r] = c;
        hnew[p][r] = og * tanhf(c);
      }
    }
    __syncthreads();            // all waves done reading old h
#pragma unroll
    for (int p = 0; p < 2; ++p) {
      const int n = (nt0 + p) * 16 + lc;
#pragma unroll
      for (int r = 0; r < 8; ++r) {
        const int m = mt * 16 + hi8 + r;        // m == batch index
        _Float16 hv = (_Float16)hnew[p][r];
        h_lds[m * H_ + n] = hv;
        hs[((size_t)s * B_ + m) * H_ + n] = hv;
        if (t == S_ - 1) hT[m * H_ + n] = hnew[p][r];
      }
    }
    __syncthreads();            // h update visible before next step's reads
  }
}

// ---------------- attention scores: scores[b,s,t] = <out[s,b,:], out[t,b,:]>/sqrt(512)
__global__ __launch_bounds__(256) void attn_scores_kernel(
    const _Float16* __restrict__ hsf, const _Float16* __restrict__ hsb,
    float* __restrict__ scores)
{
  const int lane = threadIdx.x & 31;
  const int wave = threadIdx.x >> 5;
  const int tile = blockIdx.x * (blockDim.x >> 5) + wave;  // b*256 + st*16 + tt
  const int b  = tile >> 8;
  const int st = (tile >> 4) & 15;
  const int tt = tile & 15;
  const int lda = B_ * H_;                                 // row stride over s

  v8f acc = {0.f,0.f,0.f,0.f,0.f,0.f,0.f,0.f};
  for (int k0 = 0; k0 < D2_; k0 += 32) {
    const _Float16* base = (k0 < H_) ? (hsf + b * H_ + k0)
                                     : (hsb + b * H_ + (k0 - H_));
    v16h a  = load_a_frag(base, lda, st * 16, 0);
    v16h bb = load_b_frag(base, lda, tt * 16, 0);   // B = A^T -> same row reads
    acc = wmma32(a, bb, acc);
  }
  const float scale = 0.04419417382415922f;  // 1/sqrt(512)
  const int tcol  = tt * 16 + (lane & 15);
  const int sbase = st * 16 + ((lane >> 4) << 3);
#pragma unroll
  for (int r = 0; r < 8; ++r)
    scores[((size_t)b << 16) + (size_t)(sbase + r) * S_ + tcol] = acc[r] * scale;
}

// ---------------- small VALU kernels ----------------
__global__ void cvt_f32_f16_kernel(const float* __restrict__ in,
                                   _Float16* __restrict__ out, int n) {
  int i = blockIdx.x * blockDim.x + threadIdx.x;
  if (i < n) out[i] = (_Float16)in[i];
}

// softmax over s (axis=1) per (b,t) column
__global__ void softmax_s_kernel(float* __restrict__ sc) {
  const int b = blockIdx.x >> 8, t = blockIdx.x & 255, s = threadIdx.x;
  const size_t idx = ((size_t)b << 16) + (size_t)s * S_ + t;
  float v = sc[idx];
  __shared__ float red[256];
  red[s] = v; __syncthreads();
  for (int o = 128; o; o >>= 1) { if (s < o) red[s] = fmaxf(red[s], red[s + o]); __syncthreads(); }
  const float mx = red[0]; __syncthreads();
  const float e = __expf(v - mx);
  red[s] = e; __syncthreads();
  for (int o = 128; o; o >>= 1) { if (s < o) red[s] += red[s + o]; __syncthreads(); }
  sc[idx] = e / red[0];
}

__global__ void rowsum_kernel(const float* __restrict__ a, float* __restrict__ rs) {
  const int b = blockIdx.x >> 8, s = blockIdx.x & 255, t = threadIdx.x;
  __shared__ float red[256];
  red[t] = a[((size_t)b << 16) + (size_t)s * S_ + t];
  __syncthreads();
  for (int o = 128; o; o >>= 1) { if (t < o) red[t] += red[t + o]; __syncthreads(); }
  if (t == 0) rs[b * S_ + s] = red[0];
}

// cumulative sum over the batch axis (the reference's "cumulative concat bug")
__global__ void cumsum_b_kernel(const float* __restrict__ rs, float* __restrict__ cum) {
  const int s = threadIdx.x;
  float acc = 0.f;
  for (int b = 0; b < B_; ++b) { acc += rs[b * S_ + s]; cum[b * S_ + s] = acc; }
}

__global__ void cdet_kernel(const _Float16* __restrict__ hsf, const _Float16* __restrict__ hsb,
                            const float* __restrict__ cum, _Float16* __restrict__ out) {
  const int i = blockIdx.x * blockDim.x + threadIdx.x;   // S*B*512
  if (i >= S_ * B_ * D2_) return;
  const int d = i & 511, sb = i >> 9, b = sb & 63, s = sb >> 6;
  const float v = (float)(d < H_ ? hsf[(size_t)sb * H_ + d] : hsb[(size_t)sb * H_ + d - H_]);
  out[i] = (_Float16)(v * cum[b * S_ + s]);
}

__global__ void h1_kernel(const float* __restrict__ hTf, const float* __restrict__ hTb,
                          _Float16* __restrict__ h1) {
  const int i = blockIdx.x * blockDim.x + threadIdx.x;   // B*512
  if (i >= B_ * D2_) return;
  const int b = i >> 9, d = i & 511;
  const float v = d < H_ ? hTf[b * H_ + d] : hTb[b * H_ + d - H_];
  h1[i] = (_Float16)tanf(v);            // torch.tan, not tanh
}

__global__ void reparam_kernel(const float* __restrict__ mu, const float* __restrict__ lv,
                               const float* __restrict__ eps, _Float16* __restrict__ out, int n) {
  const int i = blockIdx.x * blockDim.x + threadIdx.x;
  if (i >= n) return;
  out[i] = (_Float16)(mu[i] + eps[i] * __expf(0.5f * softplusf_(lv[i])));
}

__global__ void zandc_kernel(const _Float16* __restrict__ z, const _Float16* __restrict__ cf,
                             _Float16* __restrict__ out) {
  const int i = blockIdx.x * blockDim.x + threadIdx.x;   // S*B*128
  if (i >= S_ * B_ * 128) return;
  const int d = i & 127, sb = i >> 7, b = sb & 63;
  out[i] = (d < LAT_) ? z[b * LAT_ + d] : cf[(size_t)sb * ATT_ + d - LAT_];
}

__global__ void tanc_kernel(const _Float16* __restrict__ df, const _Float16* __restrict__ db,
                            _Float16* __restrict__ out) {
  const int i = blockIdx.x * blockDim.x + threadIdx.x;   // S*B*512
  if (i >= S_ * B_ * D2_) return;
  const int d = i & 511, sb = i >> 9;
  const float v = (float)(d < H_ ? df[(size_t)sb * H_ + d] : db[(size_t)sb * H_ + d - H_]);
  out[i] = (_Float16)tanf(v);
}

// ---------------- host orchestration ----------------
extern "C" void kernel_launch(void* const* d_in, const int* in_sizes, int n_in,
                              void* d_out, int out_size, void* d_ws, size_t ws_size,
                              hipStream_t stream)
{
  (void)in_sizes; (void)n_in; (void)out_size; (void)ws_size;
  const float* x      = (const float*)d_in[0];
  const float* Wih_ef = (const float*)d_in[1];
  const float* Whh_ef = (const float*)d_in[2];
  const float* b_ef   = (const float*)d_in[3];
  const float* Wih_eb = (const float*)d_in[4];
  const float* Whh_eb = (const float*)d_in[5];
  const float* b_eb   = (const float*)d_in[6];
  const float* W_mu   = (const float*)d_in[7];
  const float* b_mu   = (const float*)d_in[8];
  const float* W_lv   = (const float*)d_in[9];
  const float* b_lv   = (const float*)d_in[10];
  const float* W_am   = (const float*)d_in[11];
  const float* b_am   = (const float*)d_in[12];
  const float* W_al   = (const float*)d_in[13];
  const float* b_al   = (const float*)d_in[14];
  const float* Wih_df = (const float*)d_in[15];
  const float* Whh_df = (const float*)d_in[16];
  const float* b_df   = (const float*)d_in[17];
  const float* Wih_db = (const float*)d_in[18];
  const float* Whh_db = (const float*)d_in[19];
  const float* b_db   = (const float*)d_in[20];
  const float* W_rm   = (const float*)d_in[21];
  const float* b_rm   = (const float*)d_in[22];
  const float* eps_z  = (const float*)d_in[23];
  const float* eps_c  = (const float*)d_in[24];

  char* base = (char*)d_ws; size_t off = 0;
  auto alloc = [&](size_t bytes) -> void* {
    off = (off + 255) & ~(size_t)255; void* p = base + off; off += bytes; return p;
  };
  auto af16 = [&](size_t n) { return (_Float16*)alloc(n * sizeof(_Float16)); };
  auto af32 = [&](size_t n) { return (float*)alloc(n * sizeof(float)); };

  _Float16* x_h      = af16((size_t)S_ * B_ * I_);
  _Float16* Wihef_h  = af16((size_t)G4H * I_);
  _Float16* Wiheb_h  = af16((size_t)G4H * I_);
  _Float16* Whhef_h  = af16((size_t)G4H * H_);
  _Float16* Whheb_h  = af16((size_t)G4H * H_);
  _Float16* Wihdf_h  = af16((size_t)G4H * 128);
  _Float16* Wihdb_h  = af16((size_t)G4H * 128);
  _Float16* Whhdf_h  = af16((size_t)G4H * H_);
  _Float16* Whhdb_h  = af16((size_t)G4H * H_);
  _Float16* Wmu_h    = af16((size_t)LAT_ * D2_);
  _Float16* Wlv_h    = af16((size_t)LAT_ * D2_);
  _Float16* Wam_h    = af16((size_t)ATT_ * D2_);
  _Float16* Wal_h    = af16((size_t)ATT_ * D2_);
  _Float16* Wrm_h    = af16((size_t)I_ * D2_);
  _Float16* xWf_h    = af16((size_t)S_ * B_ * G4H);   // swizzled; reused enc->dec
  _Float16* xWb_h    = af16((size_t)S_ * B_ * G4H);
  _Float16* hsf_h    = af16((size_t)S_ * B_ * H_);
  _Float16* hsb_h    = af16((size_t)S_ * B_ * H_);
  _Float16* dhsf_h   = af16((size_t)S_ * B_ * H_);
  _Float16* dhsb_h   = af16((size_t)S_ * B_ * H_);
  float*    hTf      = af32((size_t)B_ * H_);
  float*    hTb      = af32((size_t)B_ * H_);
  _Float16* h1_h     = af16((size_t)B_ * D2_);
  float*    mu_pre   = af32((size_t)B_ * LAT_);
  float*    lv_pre   = af32((size_t)B_ * LAT_);
  _Float16* z_h      = af16((size_t)B_ * LAT_);
  float*    scores   = af32((size_t)B_ * S_ * S_);
  float*    rowsum   = af32((size_t)B_ * S_);
  float*    cum      = af32((size_t)B_ * S_);
  _Float16* cdet_h   = af16((size_t)S_ * B_ * D2_);   // reused as tan(out_dec)
  float*    muc_pre  = af32((size_t)S_ * B_ * ATT_);
  float*    lvc_pre  = af32((size_t)S_ * B_ * ATT_);
  _Float16* cfin_h   = af16((size_t)S_ * B_ * ATT_);
  _Float16* zc_h     = af16((size_t)S_ * B_ * 128);

  auto cvt = [&](const float* in, _Float16* o, int n) {
    cvt_f32_f16_kernel<<<(n + 255) / 256, 256, 0, stream>>>(in, o, n);
  };
  auto gemm = [&](const _Float16* A, const _Float16* W, const float* bias, void* C,
                  int M, int N, int K, int out_mode) {
    int tiles = (M / 16) * (N / 16);
    wmma_gemm_kernel<<<(tiles + 7) / 8, 256, 0, stream>>>(A, W, bias, C, M, N, K, out_mode);
  };

  // ---- fp32 -> f16 staging ----
  cvt(x, x_h, S_ * B_ * I_);
  cvt(Wih_ef, Wihef_h, G4H * I_);   cvt(Wih_eb, Wiheb_h, G4H * I_);
  cvt(Whh_ef, Whhef_h, G4H * H_);   cvt(Whh_eb, Whheb_h, G4H * H_);
  cvt(Wih_df, Wihdf_h, G4H * 128);  cvt(Wih_db, Wihdb_h, G4H * 128);
  cvt(Whh_df, Whhdf_h, G4H * H_);   cvt(Whh_db, Whhdb_h, G4H * H_);
  cvt(W_mu, Wmu_h, LAT_ * D2_);     cvt(W_lv, Wlv_h, LAT_ * D2_);
  cvt(W_am, Wam_h, ATT_ * D2_);     cvt(W_al, Wal_h, ATT_ * D2_);
  cvt(W_rm, Wrm_h, I_ * D2_);

  // ---- encoder: xW = x@Wih^T + b (all steps, fragment-swizzled), then recurrence ----
  gemm(x_h, Wihef_h, b_ef, xWf_h, S_ * B_, G4H, I_, 2);
  gemm(x_h, Wiheb_h, b_eb, xWb_h, S_ * B_, G4H, I_, 2);
  lstm_bidir_kernel<<<2, 1024, 0, stream>>>(xWf_h, xWb_h, Whhef_h, Whheb_h,
                                            hsf_h, hsb_h, hTf, hTb);

  // ---- mu / logvar / z ----
  h1_kernel<<<(B_ * D2_ + 255) / 256, 256, 0, stream>>>(hTf, hTb, h1_h);
  gemm(h1_h, Wmu_h, b_mu, mu_pre, B_, LAT_, D2_, 0);
  gemm(h1_h, Wlv_h, b_lv, lv_pre, B_, LAT_, D2_, 0);
  reparam_kernel<<<(B_ * LAT_ + 255) / 256, 256, 0, stream>>>(mu_pre, lv_pre, eps_z, z_h, B_ * LAT_);

  // ---- attention ----
  attn_scores_kernel<<<(B_ * 16 * 16) / 8, 256, 0, stream>>>(hsf_h, hsb_h, scores);
  softmax_s_kernel<<<B_ * S_, 256, 0, stream>>>(scores);
  rowsum_kernel<<<B_ * S_, 256, 0, stream>>>(scores, rowsum);
  cumsum_b_kernel<<<1, 256, 0, stream>>>(rowsum, cum);
  cdet_kernel<<<(S_ * B_ * D2_ + 255) / 256, 256, 0, stream>>>(hsf_h, hsb_h, cum, cdet_h);

  // ---- variational attention ----
  gemm(cdet_h, Wam_h, b_am, muc_pre, S_ * B_, ATT_, D2_, 0);
  gemm(cdet_h, Wal_h, b_al, lvc_pre, S_ * B_, ATT_, D2_, 0);
  reparam_kernel<<<(S_ * B_ * ATT_ + 255) / 256, 256, 0, stream>>>(muc_pre, lvc_pre, eps_c,
                                                                   cfin_h, S_ * B_ * ATT_);

  // ---- decoder ----
  zandc_kernel<<<(S_ * B_ * 128 + 255) / 256, 256, 0, stream>>>(z_h, cfin_h, zc_h);
  gemm(zc_h, Wihdf_h, b_df, xWf_h, S_ * B_, G4H, 128, 2);
  gemm(zc_h, Wihdb_h, b_db, xWb_h, S_ * B_, G4H, 128, 2);
  lstm_bidir_kernel<<<2, 1024, 0, stream>>>(xWf_h, xWb_h, Whhdf_h, Whhdb_h,
                                            dhsf_h, dhsb_h, hTf, hTb);

  // ---- output projection: tan(out_dec) @ W_rm^T + b_rm -> d_out (f32) ----
  tanc_kernel<<<(S_ * B_ * D2_ + 255) / 256, 256, 0, stream>>>(dhsf_h, dhsb_h, cdet_h);
  gemm(cdet_h, Wrm_h, b_rm, d_out, S_ * B_, I_, D2_, 0);
}
